// LSTM_91259465105415
// MI455X (gfx1250) — matrix operand order repaired and checked by reading, not compiled, not run on hardware
//
#include <hip/hip_runtime.h>
#include <hip/hip_bf16.h>

// ---------------------------------------------------------------------------
// LSTM on CDNA5 (gfx1250): bf16 WMMA GEMMs with fp32 accumulation.
//   Phase 0: convert embeds -> bf16; transpose+convert weights -> bf16 [N,K].
//   Phase 1: x-projections for all timesteps (big parallel WMMA GEMM + bias).
//   Phase 2: 512 stream-ordered fused step kernels. 8 waves/block:
//            {4 gates} x {2 K-halves} split-K, partials combined in LDS,
//            x-projection operands prefetched into registers before the GEMM
//            so HBM latency overlaps the WMMA chain, fused sigmoid/tanh +
//            c/h update, bf16 h double-buffered across steps.
// ---------------------------------------------------------------------------

typedef __attribute__((ext_vector_type(16))) __bf16 v16bf;
typedef __attribute__((ext_vector_type(8)))  __bf16 v8bf;
typedef __attribute__((ext_vector_type(8)))  float  v8f;

#define T_SEQ 512
#define BATCH 64
#define EMB   512
#define HID   1024
#define BH    (BATCH * HID)                 // 65536
#define TBH   ((size_t)T_SEQ * (size_t)BH)  // 33554432 elements per gate

// A-matrix (16x32 bf16) lane layout used by this code (contiguous k, matching load_b_tile):
//   lanes 0-15 : K = 0..15  in v0..7   (l2 = 0)
//   lanes 16-31: K = 16..31 in v0..7   (l2 = 1)
__device__ __forceinline__ v16bf load_a_tile(const __bf16* __restrict__ row,
                                             int k0, int l2) {
  union { v16bf v; v8bf h[2]; } u;
  u.h[0] = *(const v8bf*)(row + k0 + 16 * l2);
  u.h[1] = *(const v8bf*)(row + k0 + 16 * l2 + 8);
  return u.v;
}

// B-matrix (32x16 bf16) lane layout: lanes 0-15 hold K=0..15, lanes 16-31 hold
// K=16..31 for column n = lane%16 (weights stored transposed [N,K] so this is
// one contiguous 32-byte span = two b128 loads).
__device__ __forceinline__ v16bf load_b_tile(const __bf16* __restrict__ col,
                                             int k0, int l2) {
  union { v16bf v; v8bf h[2]; } u;
  u.h[0] = *(const v8bf*)(col + k0 + 16 * l2);
  u.h[1] = *(const v8bf*)(col + k0 + 16 * l2 + 8);
  return u.v;
}

__device__ __forceinline__ v8f wmma_bf16(v16bf a, v16bf b, v8f c) {
  return __builtin_amdgcn_wmma_f32_16x16x32_bf16(false, a, false, b,
                                                 (short)0, c, false, false);
}

__device__ __forceinline__ float sigmoidf_(float x) {
  return 1.0f / (1.0f + __expf(-x));
}

// ---------------------------------------------------------------------------
// Phase 0 kernels
// ---------------------------------------------------------------------------
__global__ void cvt_bf16_kernel(const float* __restrict__ in,
                                __bf16* __restrict__ out, int n) {
  int i = blockIdx.x * blockDim.x + threadIdx.x;
  if (i < n) out[i] = (__bf16)in[i];
}

// in: [K,N] fp32 row-major  ->  out: [N,K] bf16 row-major
__global__ void transpose_bf16_kernel(const float* __restrict__ in,
                                      __bf16* __restrict__ out, int K, int N) {
  int i = blockIdx.x * blockDim.x + threadIdx.x;
  if (i < K * N) {
    int k = i / N, n = i % N;
    out[(size_t)n * K + k] = (__bf16)in[i];
  }
}

__global__ void init_state_kernel(__bf16* __restrict__ hbf,
                                  float* __restrict__ c) {
  int i = blockIdx.x * blockDim.x + threadIdx.x;
  if (i < BH) {
    hbf[i] = (__bf16)0.0f;
    c[i]   = 0.0f;
  }
}

// ---------------------------------------------------------------------------
// Phase 1: x-projection GEMM. C[M,HID] = A[M,EMB] * Bt[HID,EMB]^T + bias
// Block = 128 threads = 4 waves; wave w computes the 16x16 tile at
// (blockIdx.y*64 + w*16, blockIdx.x*16).
// ---------------------------------------------------------------------------
__global__ __launch_bounds__(128) void xproj_gemm_kernel(
    const __bf16* __restrict__ A,    // [T*B, EMB] bf16
    const __bf16* __restrict__ Bt,   // [HID, EMB] bf16 (transposed weight)
    const float* __restrict__ bias,  // [HID]
    float* __restrict__ out) {       // [T*B, HID] fp32
  const int lane   = threadIdx.x & 31;
  const int wave   = threadIdx.x >> 5;
  const int lanelo = lane & 15;
  const int l2     = lane >> 4;
  const int n0     = blockIdx.x * 16;
  const int m0     = blockIdx.y * 64 + wave * 16;

  const __bf16* arow = A + (size_t)(m0 + lanelo) * EMB;
  const __bf16* bcol = Bt + (size_t)(n0 + lanelo) * EMB;

  v8f acc = {0.f, 0.f, 0.f, 0.f, 0.f, 0.f, 0.f, 0.f};
#pragma unroll 4
  for (int k0 = 0; k0 < EMB; k0 += 32) {
    v16bf a = load_a_tile(arow, k0, l2);
    v16bf b = load_b_tile(bcol, k0, l2);
    acc = wmma_bf16(a, b, acc);
  }

  const float bv = bias[n0 + lanelo];
#pragma unroll
  for (int r = 0; r < 8; ++r) {
    out[(size_t)(m0 + r + 8 * l2) * HID + (n0 + lanelo)] = acc[r] + bv;
  }
}

// ---------------------------------------------------------------------------
// Phase 2: one fused LSTM step. Grid = HID/16 blocks, 256 threads = 8 waves.
// Wave w: gate = w&3, K-half = w>>2 (512 columns of K each). Each wave runs
// 16 k-iters x 4 M-tiles = 64 WMMAs for its 64x16 partial of h @ Wh[gate].
// Partials land in LDS, barrier, then the block fuses the gate math and the
// c/h state update for its 64x16 slice. xbuf operands (HBM-resident) are
// prefetched into registers BEFORE the GEMM so VMEM latency overlaps WMMA.
// ---------------------------------------------------------------------------
__global__ __launch_bounds__(256) void lstm_step_kernel(
    const float* __restrict__ xbuf,   // 4 gates x [T,B,H] fp32 (bias folded)
    const __bf16* __restrict__ Wh,    // 4 gates x [HID,HID] bf16 transposed
    const __bf16* __restrict__ hin,   // [B,H] bf16 (prev h)
    __bf16* __restrict__ hout,        // [B,H] bf16 (next h)
    float* __restrict__ c,            // [B,H] fp32 cell state
    float* __restrict__ out,          // [T,B,H] fp32
    int t) {
  __shared__ float pre[8][64][16];  // 32 KB: [gate + 4*khalf][m][col]

  const int tid    = threadIdx.x;
  const int lane   = tid & 31;
  const int wave   = tid >> 5;
  const int gate   = wave & 3;  // 0=i,1=f,2=g,3=o
  const int khalf  = wave >> 2; // 0: K=[0,512)  1: K=[512,1024)
  const int lanelo = lane & 15;
  const int l2     = lane >> 4;
  const int n0     = blockIdx.x * 16;

  // ---- Prefetch x-projections + cell state for this thread's 4 elements ----
  const size_t base = (size_t)t * (size_t)BH;
  float xi[4], xf[4], xg[4], xo[4], cold[4];
#pragma unroll
  for (int e = 0; e < 4; ++e) {
    int lin = e * 256 + tid;            // 0..1023 over the 64x16 slice
    int m   = lin >> 4;
    int col = lin & 15;
    size_t sidx = (size_t)m * HID + (n0 + col);
    size_t gidx = base + sidx;
    xi[e]   = xbuf[0 * TBH + gidx];
    xf[e]   = xbuf[1 * TBH + gidx];
    xg[e]   = xbuf[2 * TBH + gidx];
    xo[e]   = xbuf[3 * TBH + gidx];
    cold[e] = c[sidx];
  }

  // ---- Split-K GEMM: this wave's 64x16 partial over K-half ----
  const int kbase = khalf * (HID / 2);
  const __bf16* bcol =
      Wh + (size_t)gate * HID * HID + (size_t)(n0 + lanelo) * HID + kbase;
  const __bf16* arow = hin + (size_t)lanelo * HID + kbase;

  v8f acc[4];
#pragma unroll
  for (int mt = 0; mt < 4; ++mt)
    acc[mt] = (v8f){0.f, 0.f, 0.f, 0.f, 0.f, 0.f, 0.f, 0.f};

#pragma unroll 2
  for (int k0 = 0; k0 < HID / 2; k0 += 32) {
    v16bf b = load_b_tile(bcol, k0, l2);
#pragma unroll
    for (int mt = 0; mt < 4; ++mt) {
      v16bf a = load_a_tile(arow + (size_t)(mt * 16) * HID, k0, l2);
      acc[mt] = wmma_bf16(a, b, acc[mt]);
    }
  }

#pragma unroll
  for (int mt = 0; mt < 4; ++mt) {
#pragma unroll
    for (int r = 0; r < 8; ++r) {
      pre[wave][mt * 16 + r + 8 * l2][lanelo] = acc[mt][r];
    }
  }
  __syncthreads();

  // ---- Fused gate math + state update (combine the two K-half partials) ----
#pragma unroll
  for (int e = 0; e < 4; ++e) {
    int lin = e * 256 + tid;
    int m   = lin >> 4;
    int col = lin & 15;
    size_t sidx = (size_t)m * HID + (n0 + col);
    size_t gidx = base + sidx;

    float pi = xi[e] + pre[0][m][col] + pre[4][m][col];
    float pf = xf[e] + pre[1][m][col] + pre[5][m][col];
    float pg = xg[e] + pre[2][m][col] + pre[6][m][col];
    float po = xo[e] + pre[3][m][col] + pre[7][m][col];

    float ig = sigmoidf_(pi);
    float fg = sigmoidf_(pf);
    float gg = tanhf(pg);
    float og = sigmoidf_(po);

    float cn = gg * ig + cold[e] * fg;
    float hn = tanhf(cn) * og;

    c[sidx]    = cn;
    out[gidx]  = hn;
    hout[sidx] = (__bf16)hn;
  }
}

// ---------------------------------------------------------------------------
// Host launch
// ---------------------------------------------------------------------------
extern "C" void kernel_launch(void* const* d_in, const int* in_sizes, int n_in,
                              void* d_out, int out_size, void* d_ws,
                              size_t ws_size, hipStream_t stream) {
  (void)in_sizes; (void)n_in; (void)out_size; (void)ws_size;

  const float* embeds = (const float*)d_in[0];
  // Gate order i, f, g, o per the reference update equations.
  const float* W_x[4] = {(const float*)d_in[1], (const float*)d_in[4],
                         (const float*)d_in[7], (const float*)d_in[10]};
  const float* W_h[4] = {(const float*)d_in[2], (const float*)d_in[5],
                         (const float*)d_in[8], (const float*)d_in[11]};
  const float* bias[4] = {(const float*)d_in[3], (const float*)d_in[6],
                          (const float*)d_in[9], (const float*)d_in[12]};
  float* out = (float*)d_out;

  // Workspace carve-up.
  char* ws = (char*)d_ws;
  size_t off = 0;
  __bf16* emb_bf = (__bf16*)(ws + off); off += (size_t)T_SEQ * BATCH * EMB * 2;
  __bf16* wx_bf  = (__bf16*)(ws + off); off += (size_t)4 * HID * EMB * 2;
  __bf16* wh_bf  = (__bf16*)(ws + off); off += (size_t)4 * HID * HID * 2;
  float*  xbuf   = (float*)(ws + off);  off += (size_t)4 * TBH * 4;
  __bf16* hbufA  = (__bf16*)(ws + off); off += (size_t)BH * 2;
  __bf16* hbufB  = (__bf16*)(ws + off); off += (size_t)BH * 2;
  float*  cbuf   = (float*)(ws + off);  off += (size_t)BH * 4;

  // Phase 0: conversions / transposes / state init.
  {
    int n = T_SEQ * BATCH * EMB;
    cvt_bf16_kernel<<<(n + 255) / 256, 256, 0, stream>>>(embeds, emb_bf, n);
  }
  for (int g = 0; g < 4; ++g) {
    int n = EMB * HID;
    transpose_bf16_kernel<<<(n + 255) / 256, 256, 0, stream>>>(
        W_x[g], wx_bf + (size_t)g * HID * EMB, EMB, HID);
  }
  for (int g = 0; g < 4; ++g) {
    int n = HID * HID;
    transpose_bf16_kernel<<<(n + 255) / 256, 256, 0, stream>>>(
        W_h[g], wh_bf + (size_t)g * HID * HID, HID, HID);
  }
  init_state_kernel<<<(BH + 255) / 256, 256, 0, stream>>>(hbufA, cbuf);

  // Phase 1: x-projections for all timesteps (bias folded in).
  {
    dim3 grid(HID / 16, (T_SEQ * BATCH) / 64);
    for (int g = 0; g < 4; ++g) {
      xproj_gemm_kernel<<<grid, 128, 0, stream>>>(
          emb_bf, wx_bf + (size_t)g * HID * EMB, bias[g],
          xbuf + (size_t)g * TBH);
    }
  }

  // Phase 2: sequential recurrence, stream-ordered; double-buffered bf16 h.
  for (int t = 0; t < T_SEQ; ++t) {
    const __bf16* hin = (t & 1) ? hbufB : hbufA;
    __bf16* hout      = (t & 1) ? hbufA : hbufB;
    lstm_step_kernel<<<HID / 16, 256, 0, stream>>>(xbuf, wh_bf, hin, hout,
                                                   cbuf, out, t);
  }
}
